// AfmoeMoE_43963285242504
// MI455X (gfx1250) — matrix operand order repaired and checked by reading, not compiled, not run on hardware
//
#include <hip/hip_runtime.h>
#include <hip/hip_bf16.h>

// ---------------------------------------------------------------------------
// AFMoE MoE layer for gfx1250 (MI455X).  B=2,S=512,D=1024,E=16,K=8,I=1024.
// Dense expert evaluation (matches reference: unselected experts weight 0).
// Weights & activations pre-converted to bf16 in workspace once, so the hot
// GEMM loops are pure {global_load_b128 / ds_load_b128 -> v_wmma_f32_16x16x32_bf16}.
// bf16 weight set (~102 MB) is L2-resident (192 MB), giving cross-tile reuse.
// ---------------------------------------------------------------------------

typedef unsigned short u16;
typedef unsigned long long u64;
typedef __attribute__((ext_vector_type(16))) u16     v16u;
typedef __attribute__((ext_vector_type(8)))  u16     v8u;
typedef __attribute__((ext_vector_type(16))) __bf16  v16bf;
typedef __attribute__((ext_vector_type(8)))  float   v8f;

#define D_DIM 1024
#define I_DIM 1024
#define NEXP  16
#define TOPK  8
#define NTOK  1024          // B*S
#define LDW   1032          // LDS row stride (u16 elems), %8==0 -> 16B aligned
#define EXP_W ((size_t)I_DIM * D_DIM)   // 1M elems per expert matrix

// round-to-nearest-even fp32 -> bf16 (bits)
__device__ __forceinline__ u16 f2bf(float f) {
    unsigned int u = __builtin_bit_cast(unsigned int, f);
    u += 0x7FFFu + ((u >> 16) & 1u);
    return (u16)(u >> 16);
}

// fast silu: x * sigmoid(x) using v_exp_f32 + v_rcp_f32 (no precise-div chain)
__device__ __forceinline__ float fast_silu_mul(float g, float u) {
    return (g * u) * __builtin_amdgcn_rcpf(1.f + __expf(-g));
}

// A fragment (16x32 bf16, ISA 7.12.2): lane holds row m=lane&15;
// low lanes: K = k0+0..7 and k0+16..23 ; high lanes: +8 on both groups.
// p points at &row[m][k0 + (hi?8:0)]; reads p[0..7] and p[16..23].
__device__ __forceinline__ v16bf frag_a(const u16* p) {
    v8u g0 = *(const v8u*)(p);
    v8u g1 = *(const v8u*)(p + 16);
    v16u t;
#pragma unroll
    for (int i = 0; i < 8; ++i) { t[i] = g0[i]; t[8 + i] = g1[i]; }
    return __builtin_bit_cast(v16bf, t);
}

// B fragment (32x16 bf16): lane holds column n=lane&15; low lanes K=k0+0..7 and
// k0+16..23, high lanes +8 on both.  p points at &W[row n][k0 + (hi?8:0)]; reads p[0..7], p[16..23].
__device__ __forceinline__ v16bf frag_b(const u16* p) {
    v8u g0 = *(const v8u*)(p);
    v8u g1 = *(const v8u*)(p + 16);
    v16u t;
#pragma unroll
    for (int i = 0; i < 8; ++i) { t[i] = g0[i]; t[8 + i] = g1[i]; }
    return __builtin_bit_cast(v16bf, t);
}

// ---------------------------------------------------------------------------
// fp32 -> bf16 conversion pass (4 elems / thread, packed 8-byte stores)
// ---------------------------------------------------------------------------
__global__ __launch_bounds__(256) void cvt_kernel(const float* __restrict__ src,
                                                  u16* __restrict__ dst, int n4) {
    int i = blockIdx.x * blockDim.x + threadIdx.x;
    if (i < n4) {
        float4 f = ((const float4*)src)[i];
        u64 p = (u64)f2bf(f.x) | ((u64)f2bf(f.y) << 16) |
                ((u64)f2bf(f.z) << 32) | ((u64)f2bf(f.w) << 48);
        ((u64*)dst)[i] = p;
    }
}

__global__ void zero_kernel(float* __restrict__ p, int n) {
    int i = blockIdx.x * blockDim.x + threadIdx.x;
    if (i < n) p[i] = 0.f;
}

// ---------------------------------------------------------------------------
// Routing: sigmoid gates, top-8 (+bias), normalize, scale; dense combine[T,16]
// One 256-thread block per 16 tokens; thread = (token_local, expert).
// ---------------------------------------------------------------------------
__global__ __launch_bounds__(256) void route_kernel(
    const float* __restrict__ x, const float* __restrict__ gw,
    const float* __restrict__ bias, float* __restrict__ cw) {
    __shared__ float ssc[16][16];
    __shared__ float ssel[16][16];
    const int tid = threadIdx.x;
    const int tl = tid >> 4, e = tid & 15;
    const int t = blockIdx.x * 16 + tl;

    const float4* xr = (const float4*)(x + (size_t)t * D_DIM);
    const float4* wr = (const float4*)(gw + (size_t)e * D_DIM);
    float acc = 0.f;
    for (int i = 0; i < D_DIM / 4; ++i) {
        float4 a = xr[i], b = wr[i];
        acc += a.x * b.x + a.y * b.y + a.z * b.z + a.w * b.w;
    }
    float sc = __builtin_amdgcn_rcpf(1.f + __expf(-acc));   // sigmoid
    ssc[tl][e]  = sc;
    ssel[tl][e] = sc + bias[e];
    __syncthreads();

    if (e == 0) {
        int   idx[TOPK];
        float val[TOPK];
        float sum = 0.f;
#pragma unroll
        for (int k = 0; k < TOPK; ++k) {
            int best = 0; float bv = -1e30f;
#pragma unroll
            for (int j = 0; j < NEXP; ++j) {
                float v = ssel[tl][j];
                if (v > bv) { bv = v; best = j; }
            }
            ssel[tl][best] = -1e30f;
            idx[k] = best; val[k] = ssc[tl][best]; sum += val[k];
        }
        float scale = 2.826f * __builtin_amdgcn_rcpf(sum);
        float out[NEXP];
#pragma unroll
        for (int j = 0; j < NEXP; ++j) out[j] = 0.f;
#pragma unroll
        for (int k = 0; k < TOPK; ++k) out[idx[k]] = val[k] * scale;
        float* dst = cw + (size_t)t * NEXP;
#pragma unroll
        for (int j = 0; j < NEXP; ++j) dst[j] = out[j];
    }
}

// ---------------------------------------------------------------------------
// Expert SwiGLU MLP.  Grid: (64 token tiles, 17 experts); 256 thr = 8 waves.
// Expert 16 = shared expert (combine weight 1.0); all weights bf16 in ws.
// ---------------------------------------------------------------------------
__global__ __launch_bounds__(256) void expert_kernel(
    const u16* __restrict__ xbf, const float* __restrict__ cw,
    const u16* __restrict__ wg_all, const u16* __restrict__ wu_all,
    const u16* __restrict__ wd_all,
    float* __restrict__ y) {
    __shared__ alignas(16) u16 sX[16 * LDW];   // x tile, bf16
    __shared__ alignas(16) u16 sH[16 * LDW];   // silu(g)*u tile, bf16
    __shared__ float sW[16];                   // per-token combine weight

    const int e   = blockIdx.y;
    const int t0  = blockIdx.x * 16;
    const int tid = threadIdx.x;

    const u16* Wg = wg_all + (size_t)e * EXP_W;
    const u16* Wu = wu_all + (size_t)e * EXP_W;
    const u16* Wd = wd_all + (size_t)e * EXP_W;

    // ---- stage x tile (bf16) into LDS: 2048 16-byte vectors, 8 per thread ----
    {
        const v8u* src = (const v8u*)(xbf + (size_t)t0 * D_DIM);
#pragma unroll
        for (int j = 0; j < 8; ++j) {
            int idx = tid + 256 * j;          // 0..2047
            int row = idx >> 7;               // 128 v8u per row
            int c8  = idx & 127;
            *(v8u*)(sX + (size_t)row * LDW + c8 * 8) = src[row * 128 + c8];
        }
    }
    if (tid < 16)
        sW[tid] = (e == NEXP) ? 1.0f : cw[(size_t)(t0 + tid) * NEXP + e];
    __syncthreads();

    const int wave   = tid >> 5;
    const int lane   = tid & 31;
    const int n_lane = lane & 15;
    const int hi     = lane >> 4;
    const int kb_a   = hi << 3;   // +8 elem offset for A frag on high lanes
    const int kb_b   = hi << 3;   // +8 elem offset for B frag on high lanes

    // ---- phase 1: h = silu(x Wg^T) * (x Wu^T), 128 channels per wave ----
    const u16* aX = sX + (size_t)(lane & 15) * LDW + kb_a;
#pragma unroll 1
    for (int nt = 0; nt < 8; ++nt) {
        const int i0 = wave * 128 + nt * 16;
        const u16* gr = Wg + (size_t)(i0 + n_lane) * D_DIM + kb_b;
        const u16* ur = Wu + (size_t)(i0 + n_lane) * D_DIM + kb_b;
        v8f accg = {0.f, 0.f, 0.f, 0.f, 0.f, 0.f, 0.f, 0.f};
        v8f accu = {0.f, 0.f, 0.f, 0.f, 0.f, 0.f, 0.f, 0.f};
#pragma unroll 4
        for (int ks = 0; ks < D_DIM / 32; ++ks) {
            const int k0 = ks * 32;
            v16bf a  = frag_a(aX + k0);
            v16bf bg = frag_b(gr + k0);
            accg = __builtin_amdgcn_wmma_f32_16x16x32_bf16(
                       false, a, false, bg, (short)0, accg, false, false);
            v16bf bu = frag_b(ur + k0);
            accu = __builtin_amdgcn_wmma_f32_16x16x32_bf16(
                       false, a, false, bu, (short)0, accu, false, false);
        }
#pragma unroll
        for (int r = 0; r < 8; ++r) {
            float h = fast_silu_mul(accg[r], accu[r]);  // silu(g)*u
            int m = r + (hi << 3);                      // C/D layout row
            sH[(size_t)m * LDW + i0 + n_lane] = f2bf(h);
        }
    }
    __syncthreads();

    // ---- phase 2: y += combine * (h Wd^T), 128 outputs per wave ----
    const u16* aH = sH + (size_t)(lane & 15) * LDW + kb_a;
#pragma unroll 1
    for (int nt = 0; nt < 8; ++nt) {
        const int d0 = wave * 128 + nt * 16;
        const u16* dr = Wd + (size_t)(d0 + n_lane) * I_DIM + kb_b;
        v8f acc = {0.f, 0.f, 0.f, 0.f, 0.f, 0.f, 0.f, 0.f};
#pragma unroll 4
        for (int ks = 0; ks < I_DIM / 32; ++ks) {
            const int k0 = ks * 32;
            v16bf a = frag_a(aH + k0);
            v16bf b = frag_b(dr + k0);
            acc = __builtin_amdgcn_wmma_f32_16x16x32_bf16(
                      false, a, false, b, (short)0, acc, false, false);
        }
#pragma unroll
        for (int r = 0; r < 8; ++r) {
            int m = r + (hi << 3);
            float v = acc[r] * sW[m];
            atomicAdd(y + (size_t)(t0 + m) * D_DIM + d0 + n_lane, v);
        }
    }
}

// ---------------------------------------------------------------------------
extern "C" void kernel_launch(void* const* d_in, const int* in_sizes, int n_in,
                              void* d_out, int out_size, void* d_ws, size_t ws_size,
                              hipStream_t stream) {
    (void)in_sizes; (void)n_in; (void)ws_size;
    const float* x      = (const float*)d_in[0];
    const float* gate_w = (const float*)d_in[1];
    const float* e_bias = (const float*)d_in[2];
    const float* w_gate = (const float*)d_in[3];
    const float* w_up   = (const float*)d_in[4];
    const float* w_down = (const float*)d_in[5];
    const float* sh_g   = (const float*)d_in[6];
    const float* sh_u   = (const float*)d_in[7];
    const float* sh_d   = (const float*)d_in[8];
    float* y = (float*)d_out;                  // [1024, 1024] f32

    // ---- workspace layout ----
    // [0, 64KB)            : combine weights, float [1024][16]
    // [64KB, +2MB)         : x in bf16        [1024*1024] u16
    // then 3 arrays of 17 expert matrices each, bf16 [17*1024*1024] u16
    float* cw  = (float*)d_ws;
    u16*   xbf = (u16*)((char*)d_ws + 65536);
    u16*   wg  = xbf + (size_t)NTOK * D_DIM;
    u16*   wu  = wg + 17 * EXP_W;
    u16*   wd  = wu + 17 * EXP_W;

    const int n4_x = (NTOK * D_DIM) / 4;          // 262144
    const int n4_w = (int)(NEXP * EXP_W / 4);     // 4194304
    const int n4_s = (int)(EXP_W / 4);            // 262144

    // fp32 -> bf16 conversion passes (weights once; experts 0..15 then shared=16)
    cvt_kernel<<<(n4_x + 255) / 256, 256, 0, stream>>>(x, xbf, n4_x);
    cvt_kernel<<<(n4_w + 255) / 256, 256, 0, stream>>>(w_gate, wg, n4_w);
    cvt_kernel<<<(n4_s + 255) / 256, 256, 0, stream>>>(sh_g, wg + NEXP * EXP_W, n4_s);
    cvt_kernel<<<(n4_w + 255) / 256, 256, 0, stream>>>(w_up, wu, n4_w);
    cvt_kernel<<<(n4_s + 255) / 256, 256, 0, stream>>>(sh_u, wu + NEXP * EXP_W, n4_s);
    cvt_kernel<<<(n4_w + 255) / 256, 256, 0, stream>>>(w_down, wd, n4_w);
    cvt_kernel<<<(n4_s + 255) / 256, 256, 0, stream>>>(sh_d, wd + NEXP * EXP_W, n4_s);

    zero_kernel<<<(out_size + 255) / 256, 256, 0, stream>>>(y, out_size);
    route_kernel<<<NTOK / 16, 256, 0, stream>>>(x, gate_w, e_bias, cw);

    dim3 grid(NTOK / 16, NEXP + 1);
    expert_kernel<<<grid, 256, 0, stream>>>(xbf, cw, wg, wu, wd, y);
}
